// EncodeProcessDecode_55508157333735
// MI455X (gfx1250) — matrix operand order repaired and checked by reading, not compiled, not run on hardware
//
#include <hip/hip_runtime.h>
#include <stddef.h>

#define NN      10000
#define NE      80000
#define L       128
#define NSTEPS  15
#define LN_EPS  1e-5f

typedef __bf16  bf16_t;
typedef bf16_t  v16bf __attribute__((ext_vector_type(16)));
typedef float   v8f   __attribute__((ext_vector_type(8)));
typedef float   v4f   __attribute__((ext_vector_type(4)));

// hardware bf16 convert (RNE); gfx1250 has bf16 VALU support
static __device__ __forceinline__ bf16_t f2bf(float f) {
    return (bf16_t)f;
}

// ---- gfx1250 async Global->LDS staging (ASYNCcnt path), with safe fallback --
#if defined(__AMDGCN__) && __has_builtin(__builtin_amdgcn_global_load_async_to_lds_b128)
#define USE_ASYNC_LDS 1
typedef int v4i_gcc __attribute__((vector_size(16)));                  // matches builtin param
typedef __attribute__((address_space(1))) v4i_gcc gas_v4i;             // global
typedef __attribute__((address_space(3))) v4i_gcc las_v4i;             // LDS
#else
#define USE_ASYNC_LDS 0
#endif

static __device__ __forceinline__ void wait_async0() {
#if __has_builtin(__builtin_amdgcn_s_wait_asynccnt)
    __builtin_amdgcn_s_wait_asynccnt(0);
#else
    asm volatile("s_wait_asynccnt 0" ::: "memory");
#endif
}

// per-fragment A source pointers (concat segments)
struct Segs { const float *s0, *s1, *s2; };

template <int AMODE>
static __device__ __forceinline__ Segs make_segs(const float* A0, const float* A1,
                                                 const int* idx0, const int* idx1, int rowc)
{
    Segs sg;
    if (AMODE == 1) {
        sg.s0 = A0 + (size_t)idx0[rowc] * L;
        sg.s1 = A0 + (size_t)idx1[rowc] * L;
        sg.s2 = A1 + (size_t)rowc * L;
    } else if (AMODE == 2) {
        sg.s0 = A0 + (size_t)rowc * L;
        sg.s1 = A1 + (size_t)rowc * L;
        sg.s2 = sg.s1;
    } else {
        sg.s0 = A0 + (size_t)rowc * L;
        sg.s1 = sg.s0;
        sg.s2 = sg.s0;
    }
    return sg;
}

template <int AMODE>
static __device__ __forceinline__ const float* chunk(const Segs& sg, int k)
{
    if (AMODE == 1) return (k < 128) ? sg.s0 + k
                         : (k < 256) ? sg.s1 + (k - 128)
                                     : sg.s2 + (k - 256);
    if (AMODE == 2) return (k < 128) ? sg.s0 + k : sg.s1 + (k - 128);
    return sg.s0 + k;
}

// build 16x32 bf16 A fragment (ISA layout: lane&15 = M row,
// lane>>4 selects K 0-7|16-23 vs 8-15|24-31)
template <int AMODE>
static __device__ __forceinline__ v16bf load_a_frag(const Segs& sg, int kb)
{
    const float* p1 = chunk<AMODE>(sg, kb);
    const float* p2 = chunk<AMODE>(sg, kb + 16);
    v4f u0 = *(const v4f*)(p1);
    v4f u1 = *(const v4f*)(p1 + 4);
    v4f u2 = *(const v4f*)(p2);
    v4f u3 = *(const v4f*)(p2 + 4);
    v16bf a;
    #pragma unroll
    for (int i = 0; i < 4; ++i) {
        a[i]      = f2bf(u0[i]);
        a[i + 4]  = f2bf(u1[i]);
        a[i + 8]  = f2bf(u2[i]);
        a[i + 12] = f2bf(u3[i]);
    }
    return a;
}

// ---------------------------------------------------------------------------
// Tiled GEMM: C[M,128] = act(A[M,K] @ W[K,128] + bias), bf16 WMMA, f32 accum.
// Wb is the weight pre-converted to bf16 and pre-transposed: Wb[n*K + k].
// AMODE 0: A = A0[M,128]                           (K = 128)
// AMODE 1: A = [A0[idx0[m]] | A0[idx1[m]] | A1[m]] (K = 384)  edge-block concat
// AMODE 2: A = [A0[m] | A1[m]]                     (K = 256)  node-block concat
// Block: 128 threads (4 waves). Tile M=128 x N=128; each wave: M=32 (two 16-row
// A fragments sharing each B fragment) x N=128 (8 tiles) -> 16 WMMA per k-step.
// ---------------------------------------------------------------------------
template <int AMODE>
__global__ __launch_bounds__(128)
void gemm_wmma_bf16(const float* __restrict__ A0, const float* __restrict__ A1,
                    const int* __restrict__ idx0, const int* __restrict__ idx1,
                    const bf16_t* __restrict__ Wb, const float* __restrict__ bias,
                    float* __restrict__ C, int M, int do_relu)
{
    constexpr int K = (AMODE == 1) ? 384 : (AMODE == 2) ? 256 : 128;
    __shared__ bf16_t Wt[L * 32];              // W slab [n][k] bf16, 8 KB

    const int tid   = threadIdx.x;
    const int wave  = tid >> 5;
    const int lane  = tid & 31;
    const int nlo   = lane & 15;               // M-row (A) / N-col (B,C)
    const int khalf = lane >> 4;               // lane-group K interleave select
    const int rowbase = blockIdx.x * 128 + wave * 32;
    const int row0 = rowbase + nlo;            // fragment 0 rows
    const int row1 = rowbase + 16 + nlo;       // fragment 1 rows
    const int rowc0 = (row0 < M) ? row0 : (M - 1);   // clamp: keep EXEC all-ones
    const int rowc1 = (row1 < M) ? row1 : (M - 1);

    const Segs sg0 = make_segs<AMODE>(A0, A1, idx0, idx1, rowc0);
    const Segs sg1 = make_segs<AMODE>(A0, A1, idx0, idx1, rowc1);

    v8f acc0[8] = {};
    v8f acc1[8] = {};

    for (int k0 = 0; k0 < K; k0 += 32) {
        __syncthreads();
        // stage Wb[n=tid][k0:k0+32] (64 contiguous bytes) into LDS
        {
            const bf16_t* src = Wb + (size_t)tid * K + k0;
#if USE_ASYNC_LDS
            gas_v4i* g = (gas_v4i*)src;
            las_v4i* l = (las_v4i*)&Wt[tid * 32];
            __builtin_amdgcn_global_load_async_to_lds_b128(g + 0, l + 0, 0, 0);
            __builtin_amdgcn_global_load_async_to_lds_b128(g + 1, l + 1, 0, 0);
            __builtin_amdgcn_global_load_async_to_lds_b128(g + 2, l + 2, 0, 0);
            __builtin_amdgcn_global_load_async_to_lds_b128(g + 3, l + 3, 0, 0);
            wait_async0();
#else
            const uint4* s4 = (const uint4*)src;
            uint4* d4 = (uint4*)&Wt[tid * 32];
            d4[0] = s4[0]; d4[1] = s4[1]; d4[2] = s4[2]; d4[3] = s4[3];
#endif
        }
        __syncthreads();

        const int kb = k0 + khalf * 8;
        const v16bf a0 = load_a_frag<AMODE>(sg0, kb);
        const v16bf a1 = load_a_frag<AMODE>(sg1, kb);

        // B fragments 32x16: lane(0-15)=col N, khalf selects K 0-7|16-23 vs 8-15|24-31;
        // each B fragment feeds both A fragments.
        #pragma unroll
        for (int tn = 0; tn < 8; ++tn) {
            const bf16_t* bp = &Wt[(tn * 16 + nlo) * 32 + khalf * 8];
            v16bf b;
            #pragma unroll
            for (int i = 0; i < 8; ++i) { b[i] = bp[i]; b[i + 8] = bp[16 + i]; }
            acc0[tn] = __builtin_amdgcn_wmma_f32_16x16x32_bf16(
                false, a0, false, b, (short)0, acc0[tn], false, false);
            acc1[tn] = __builtin_amdgcn_wmma_f32_16x16x32_bf16(
                false, a1, false, b, (short)0, acc1[tn], false, false);
        }
    }

    // C/D layout: lanes 0-15: N=lane, M=r; lanes 16-31: N=lane-16, M=8+r
    const int mhi = khalf * 8;
    #pragma unroll
    for (int tn = 0; tn < 8; ++tn) {
        const int col = tn * 16 + nlo;
        const float bv = bias[col];
        #pragma unroll
        for (int r = 0; r < 8; ++r) {
            const int g0 = rowbase + mhi + r;
            float v0 = acc0[tn][r] + bv;
            if (do_relu) v0 = v0 > 0.f ? v0 : 0.f;
            if (g0 < M) C[(size_t)g0 * L + col] = v0;

            const int g1 = rowbase + 16 + mhi + r;
            float v1 = acc1[tn][r] + bv;
            if (do_relu) v1 = v1 > 0.f ? v1 : 0.f;
            if (g1 < M) C[(size_t)g1 * L + col] = v1;
        }
    }
}

// --- weight prep: W[k*128+n] (fp32) -> Wb[n*K+k] (bf16, transposed) ---------
__global__ void convert_w_t(const float* __restrict__ W, bf16_t* __restrict__ Wb, int K)
{
    const int t = blockIdx.x * blockDim.x + threadIdx.x;
    if (t >= K * L) return;
    const int k = t / L, n = t - k * L;          // coalesced read over n
    Wb[(size_t)n * K + k] = f2bf(W[t]);
}

// --- small/ragged-K dense layer (encoder K=12/7, decoder N=3) ---------------
__global__ void dense_small(const float* __restrict__ A, const float* __restrict__ W,
                            const float* __restrict__ bias, float* __restrict__ C,
                            int M, int K, int N, int do_relu)
{
    const int t = blockIdx.x * blockDim.x + threadIdx.x;
    if (t >= M * N) return;
    const int row = t / N, col = t - row * N;
    float s = bias[col];
    for (int k = 0; k < K; ++k)
        s += A[(size_t)row * K + k] * W[(size_t)k * N + col];
    if (do_relu && s < 0.f) s = 0.f;
    C[t] = s;
}

// --- LayerNorm over 128, one wave32 per row; optional residual accumulate ---
__global__ __launch_bounds__(256)
void ln_kernel(const float* __restrict__ x, const float* __restrict__ g,
               const float* __restrict__ b, float* __restrict__ out,
               float* __restrict__ carry, int M)
{
    const int wid  = threadIdx.x >> 5;
    const int lane = threadIdx.x & 31;
    const int row  = blockIdx.x * 8 + wid;
    if (row >= M) return;                       // uniform per wave: safe
    const float* xr = x + (size_t)row * L;
    float v[4], s = 0.f, s2 = 0.f;
    #pragma unroll
    for (int j = 0; j < 4; ++j) {
        v[j] = xr[lane + 32 * j];
        s += v[j]; s2 += v[j] * v[j];
    }
    #pragma unroll
    for (int off = 16; off > 0; off >>= 1) {
        s  += __shfl_xor(s,  off, 32);
        s2 += __shfl_xor(s2, off, 32);
    }
    const float mean = s * (1.f / L);
    float var = s2 * (1.f / L) - mean * mean;
    var = var < 0.f ? 0.f : var;
    const float rs = rsqrtf(var + LN_EPS);
    #pragma unroll
    for (int j = 0; j < 4; ++j) {
        const int c = lane + 32 * j;
        const float y = (v[j] - mean) * rs * g[c] + b[c];
        if (out)   out[(size_t)row * L + c] = y;
        if (carry) carry[(size_t)row * L + c] += y;
    }
}

__global__ void zero_kernel(float* __restrict__ p, int n)
{
    int t = blockIdx.x * blockDim.x + threadIdx.x;
    if (t < n) p[t] = 0.f;
}

// --- segment_sum(new_e, receivers) via f32 global atomics -------------------
__global__ void scatter_add_kernel(const float* __restrict__ e, const int* __restrict__ recv,
                                   float* __restrict__ agg, int E)
{
    int t = blockIdx.x * blockDim.x + threadIdx.x;
    if (t >= E * L) return;
    int ei = t >> 7, c = t & (L - 1);
    atomicAdd(&agg[(size_t)recv[ei] * L + c], e[t]);
}

// ---------------------------------------------------------------------------
extern "C" void kernel_launch(void* const* d_in, const int* in_sizes, int n_in,
                              void* d_out, int out_size, void* d_ws, size_t ws_size,
                              hipStream_t stream)
{
    (void)in_sizes; (void)n_in; (void)out_size; (void)ws_size;

    const float* node_f    = (const float*)d_in[0];
    const float* edge_f    = (const float*)d_in[1];
    const int*   senders   = (const int*)d_in[2];
    const int*   receivers = (const int*)d_in[3];
    const float* enW0 = (const float*)d_in[4],  *enb0 = (const float*)d_in[5];
    const float* enW1 = (const float*)d_in[6],  *enb1 = (const float*)d_in[7];
    const float* enW2 = (const float*)d_in[8],  *enb2 = (const float*)d_in[9];
    const float* eng  = (const float*)d_in[10], *enbe = (const float*)d_in[11];
    const float* eeW0 = (const float*)d_in[12], *eeb0 = (const float*)d_in[13];
    const float* eeW1 = (const float*)d_in[14], *eeb1 = (const float*)d_in[15];
    const float* eeW2 = (const float*)d_in[16], *eeb2 = (const float*)d_in[17];
    const float* eeg  = (const float*)d_in[18], *eebe = (const float*)d_in[19];
    const float* beW0 = (const float*)d_in[20], *beb0 = (const float*)d_in[21];
    const float* beW1 = (const float*)d_in[22], *beb1 = (const float*)d_in[23];
    const float* beW2 = (const float*)d_in[24], *beb2 = (const float*)d_in[25];
    const float* beg  = (const float*)d_in[26], *bebe = (const float*)d_in[27];
    const float* bnW0 = (const float*)d_in[28], *bnb0 = (const float*)d_in[29];
    const float* bnW1 = (const float*)d_in[30], *bnb1 = (const float*)d_in[31];
    const float* bnW2 = (const float*)d_in[32], *bnb2 = (const float*)d_in[33];
    const float* bng  = (const float*)d_in[34], *bnbe = (const float*)d_in[35];
    const float* dW0  = (const float*)d_in[36], *db0  = (const float*)d_in[37];
    const float* dW1  = (const float*)d_in[38], *db1  = (const float*)d_in[39];
    const float* dW2  = (const float*)d_in[40], *db2  = (const float*)d_in[41];

    // ---- workspace: fp32 activations (~143 MB) + bf16 weights (~4.6 MB) ----
    float* ws  = (float*)d_ws;
    float* nl  = ws;                            // [NN,128] node latents
    float* el  = nl  + (size_t)NN * L;          // [NE,128] edge latents
    float* eh1 = el  + (size_t)NE * L;          // [NE,128] edge temp A
    float* eh2 = eh1 + (size_t)NE * L;          // [NE,128] edge temp B / new_e
    float* agg = eh2 + (size_t)NE * L;          // [NN,128]
    float* nh1 = agg + (size_t)NN * L;          // [NN,128]
    float* nh2 = nh1 + (size_t)NN * L;          // [NN,128]

    const size_t S1 = (size_t)L * 128;          // 16384
    const size_t S2 = (size_t)L * 256;          // 32768
    const size_t S3 = (size_t)L * 384;          // 49152
    bf16_t* wb    = (bf16_t*)(nh2 + (size_t)NN * L);
    bf16_t* enW1t = wb;
    bf16_t* enW2t = enW1t + S1;
    bf16_t* eeW1t = enW2t + S1;
    bf16_t* eeW2t = eeW1t + S1;
    bf16_t* dW0t  = eeW2t + S1;
    bf16_t* dW1t  = dW0t  + S1;
    bf16_t* beW0t = dW1t  + S1;                 // 15 x S3
    bf16_t* beW1t = beW0t + 15 * S3;            // 15 x S1
    bf16_t* beW2t = beW1t + 15 * S1;            // 15 x S1
    bf16_t* bnW0t = beW2t + 15 * S1;            // 15 x S2
    bf16_t* bnW1t = bnW0t + 15 * S2;            // 15 x S1
    bf16_t* bnW2t = bnW1t + 15 * S1;            // 15 x S1

    const dim3 b128(128), b256(256);
    const int gN  = (NN + 127) / 128;
    const int gE  = (NE + 127) / 128;
    const int lnN = (NN + 7) / 8;
    const int lnE = (NE + 7) / 8;
    const int tN  = (NN * L + 255) / 256;
    const int tE  = (NE * L + 255) / 256;

    // ---------------- weight prep (bf16, transposed) ----------------
    auto cvt = [&](const float* W, bf16_t* Wb, int K) {
        convert_w_t<<<(K * L + 255) / 256, b256, 0, stream>>>(W, Wb, K);
    };
    cvt(enW1, enW1t, 128);  cvt(enW2, enW2t, 128);
    cvt(eeW1, eeW1t, 128);  cvt(eeW2, eeW2t, 128);
    cvt(dW0, dW0t, 128);    cvt(dW1, dW1t, 128);
    for (int s = 0; s < NSTEPS; ++s) {
        cvt(beW0 + (size_t)s * 384 * L, beW0t + (size_t)s * S3, 384);
        cvt(beW1 + (size_t)s * L * L,   beW1t + (size_t)s * S1, 128);
        cvt(beW2 + (size_t)s * L * L,   beW2t + (size_t)s * S1, 128);
        cvt(bnW0 + (size_t)s * 256 * L, bnW0t + (size_t)s * S2, 256);
        cvt(bnW1 + (size_t)s * L * L,   bnW1t + (size_t)s * S1, 128);
        cvt(bnW2 + (size_t)s * L * L,   bnW2t + (size_t)s * S1, 128);
    }

    // ---------------- Encoder ----------------
    dense_small<<<tN, b256, 0, stream>>>(node_f, enW0, enb0, nh1, NN, 12, L, 1);
    gemm_wmma_bf16<0><<<gN, b128, 0, stream>>>(nh1, nullptr, nullptr, nullptr, enW1t, enb1, nh2, NN, 1);
    gemm_wmma_bf16<0><<<gN, b128, 0, stream>>>(nh2, nullptr, nullptr, nullptr, enW2t, enb2, nh1, NN, 0);
    ln_kernel<<<lnN, b256, 0, stream>>>(nh1, eng, enbe, nl, nullptr, NN);

    dense_small<<<tE, b256, 0, stream>>>(edge_f, eeW0, eeb0, eh1, NE, 7, L, 1);
    gemm_wmma_bf16<0><<<gE, b128, 0, stream>>>(eh1, nullptr, nullptr, nullptr, eeW1t, eeb1, eh2, NE, 1);
    gemm_wmma_bf16<0><<<gE, b128, 0, stream>>>(eh2, nullptr, nullptr, nullptr, eeW2t, eeb2, eh1, NE, 0);
    ln_kernel<<<lnE, b256, 0, stream>>>(eh1, eeg, eebe, el, nullptr, NE);

    // ---------------- Processor: 15 GraphNet blocks ----------------
    for (int s = 0; s < NSTEPS; ++s) {
        const float* B0 = beb0 + (size_t)s * L;
        const float* B1 = beb1 + (size_t)s * L;
        const float* B2 = beb2 + (size_t)s * L;
        const float* G  = beg  + (size_t)s * L;  const float* BE = bebe + (size_t)s * L;
        const float* nB0 = bnb0 + (size_t)s * L;
        const float* nB1 = bnb1 + (size_t)s * L;
        const float* nB2 = bnb2 + (size_t)s * L;
        const float* nG  = bng  + (size_t)s * L; const float* nBE = bnbe + (size_t)s * L;

        // edge MLP on fused concat(n[senders], n[receivers], e)  (K=384)
        gemm_wmma_bf16<1><<<gE, b128, 0, stream>>>(nl, el, senders, receivers,
                                                   beW0t + (size_t)s * S3, B0, eh1, NE, 1);
        gemm_wmma_bf16<0><<<gE, b128, 0, stream>>>(eh1, nullptr, nullptr, nullptr,
                                                   beW1t + (size_t)s * S1, B1, eh2, NE, 1);
        gemm_wmma_bf16<0><<<gE, b128, 0, stream>>>(eh2, nullptr, nullptr, nullptr,
                                                   beW2t + (size_t)s * S1, B2, eh1, NE, 0);
        // eh2 = new_e = LN(mlp);  el += new_e
        ln_kernel<<<lnE, b256, 0, stream>>>(eh1, G, BE, eh2, el, NE);

        // agg = segment_sum(new_e, receivers)
        zero_kernel<<<tN, b256, 0, stream>>>(agg, NN * L);
        scatter_add_kernel<<<tE, b256, 0, stream>>>(eh2, receivers, agg, NE);

        // node MLP on fused concat(n, agg)  (K=256)
        gemm_wmma_bf16<2><<<gN, b128, 0, stream>>>(nl, agg, nullptr, nullptr,
                                                   bnW0t + (size_t)s * S2, nB0, nh1, NN, 1);
        gemm_wmma_bf16<0><<<gN, b128, 0, stream>>>(nh1, nullptr, nullptr, nullptr,
                                                   bnW1t + (size_t)s * S1, nB1, nh2, NN, 1);
        gemm_wmma_bf16<0><<<gN, b128, 0, stream>>>(nh2, nullptr, nullptr, nullptr,
                                                   bnW2t + (size_t)s * S1, nB2, nh1, NN, 0);
        // nl += LN(mlp)
        ln_kernel<<<lnN, b256, 0, stream>>>(nh1, nG, nBE, nullptr, nl, NN);
    }

    // ---------------- Decoder ----------------
    gemm_wmma_bf16<0><<<gN, b128, 0, stream>>>(nl, nullptr, nullptr, nullptr, dW0t, db0, nh1, NN, 1);
    gemm_wmma_bf16<0><<<gN, b128, 0, stream>>>(nh1, nullptr, nullptr, nullptr, dW1t, db1, nh2, NN, 1);
    dense_small<<<(NN * 3 + 255) / 256, b256, 0, stream>>>(nh2, dW2, db2, (float*)d_out, NN, L, 3, 0);
}
